// QuantumKernelAttention_65481071395280
// MI455X (gfx1250) — matrix operand order repaired and checked by reading, not compiled, not run on hardware
//
#include <hip/hip_runtime.h>
#include <hip/hip_bf16.h>
#include <math.h>

// ---------------------------------------------------------------------------
// QuantumKernelAttention for MI455X (gfx1250, wave32, WMMA).
//
//   q/k "quantum measurement" = cos() of first NQ=8 features per head, so
//   Q/K projections only need 64 of 512 columns (8x FLOP cut), and attention
//   scores have inner rank 8. Attention is fused flash-style: the 512 MB
//   [B,H,S,S] score tensor never touches HBM (it alone would cost ~65us of
//   the 23.3 TB/s HBM budget; everything else here is ~100 MB total).
//   All matmuls run on v_wmma_f32_16x16x32_f16 (f16 in, f32 accumulate).
// ---------------------------------------------------------------------------

typedef __attribute__((ext_vector_type(16))) _Float16 v16h;
typedef __attribute__((ext_vector_type(8)))  float    v8f;

static constexpr int B  = 4;
static constexpr int S  = 2048;
static constexpr int E  = 512;
static constexpr int H  = 8;
static constexpr int DK = 64;   // E / H
static constexpr int NQ = 8;

static __device__ __forceinline__ v16h zero_v16h() {
  v16h z;
#pragma unroll
  for (int i = 0; i < 16; ++i) z[i] = (_Float16)0.0f;
  return z;
}
static __device__ __forceinline__ v8f zero_v8f() {
  v8f z;
#pragma unroll
  for (int i = 0; i < 8; ++i) z[i] = 0.0f;
  return z;
}

// ---------------------------------------------------------------------------
// Projection GEMM:  Out = epilogue( A[M,E] @ W[E,N] + bias )
//   MODE 0: A=f32 x,  N=512, out = f16 V in [B,H,S,DK] layout
//   MODE 1: A=f32 x,  N=64 selected cols (h*64+j, j<8), out = f16 cos() in [B,H,S,NQ]
//   MODE 2: A=f16 O,  N=512, out = f32 [B,S,E] (final result)
// Block: 256 threads = 8 waves. Each wave owns one 16x16 WMMA tile;
// A k-slab is staged (and f32->f16 converted) once per block in LDS.
// ---------------------------------------------------------------------------
template <int MODE>
__global__ __launch_bounds__(256) void gemm_proj(const void* __restrict__ Ain,
                                                 const float* __restrict__ W,
                                                 const float* __restrict__ bias,
                                                 void* __restrict__ Out) {
  constexpr int WN   = (MODE == 1) ? 4 : 8;   // waves along N
  constexpr int WM   = 8 / WN;                // waves along M
  constexpr int ROWS = WM * 16;

  __shared__ __align__(16) _Float16 sA[ROWS * 32];

  const int t    = threadIdx.x;
  const int wave = t >> 5;
  const int lane = t & 31;
  const int hi   = lane >> 4;   // which 16-lane half
  const int nl   = lane & 15;
  const int wm   = wave / WN;
  const int wn   = wave % WN;
  const int m0    = blockIdx.y * ROWS;
  const int mbase = m0 + wm * 16;
  const int nbase = blockIdx.x * (WN * 16) + wn * 16;

  const int ncol = nbase + nl;
  const int nsrc = (MODE == 1) ? ((ncol >> 3) * DK + (ncol & 7)) : ncol;

  v8f acc = zero_v8f();

  for (int k0 = 0; k0 < E; k0 += 32) {
    __syncthreads();
    // ---- cooperative A slab stage: ROWS x 32, coalesced, f32->f16 ----
#pragma unroll
    for (int i = 0; i < (ROWS * 32) / 256; ++i) {
      const int idx = t + i * 256;
      const int row = idx >> 5;
      const int col = idx & 31;
      float av;
      if (MODE == 2)
        av = (float)((const _Float16*)Ain)[(size_t)(m0 + row) * E + k0 + col];
      else
        av = ((const float*)Ain)[(size_t)(m0 + row) * E + k0 + col];
      sA[idx] = (_Float16)av;
    }
    __syncthreads();

    // ---- A fragment (16x32, wave32 layout: M=nl, K half by lane group) ----
    v16h af;
#pragma unroll
    for (int e = 0; e < 8; ++e)
      af[e] = sA[(wm * 16 + nl) * 32 + hi * 16 + e];
#pragma unroll
    for (int e = 8; e < 16; ++e)
      af[e] = sA[(wm * 16 + nl) * 32 + hi * 16 + e];

    // ---- B fragment (32x16: N=nl, element e -> K = k0 + hi*16 + e) ----
    v16h bf;
#pragma unroll
    for (int e = 0; e < 16; ++e)
      bf[e] = (_Float16)W[(size_t)(k0 + hi * 16 + e) * E + nsrc];

    if (k0 + 32 < E)  // hint the streaming weight slab -> global_prefetch
      __builtin_prefetch(&W[(size_t)(k0 + 32 + hi * 16) * E + nsrc], 0, 1);

    acc = __builtin_amdgcn_wmma_f32_16x16x32_f16(false, af, false, bf,
                                                 (short)0, acc, false, false);
  }

  // ---- epilogue (C/D layout: VGPR r -> M = r + 8*hi, N = nl) ----
#pragma unroll
  for (int r = 0; r < 8; ++r) {
    const int m  = mbase + r + 8 * hi;
    const int bi = m >> 11;        // m / S
    const int si = m & (S - 1);    // m % S
    if (MODE == 0) {
      const float val = acc[r] + bias[ncol];
      const int h = ncol >> 6, d = ncol & 63;
      ((_Float16*)Out)[(((size_t)(bi * H + h)) * S + si) * DK + d] = (_Float16)val;
    } else if (MODE == 1) {
      const float val = cosf(acc[r] + bias[nsrc]);
      const int h = ncol >> 3, j = ncol & 7;
      ((_Float16*)Out)[(((size_t)(bi * H + h)) * S + si) * NQ + j] = (_Float16)val;
    } else {
      ((float*)Out)[(size_t)m * E + ncol] = acc[r] + bias[ncol];
    }
  }
}

// ---------------------------------------------------------------------------
// Fused flash attention. Grid (S/128, B*H); block 256 = 8 waves.
// Each wave owns a 16-query tile; the block streams shared 32-key blocks:
//   scores: 2x wmma (rank-8 zero-padded to K=32), online softmax with
//   16-lane shuffle reductions, P bounced through wave-private LDS
//   (s_wait_dscnt) into A-fragment form, PV: 4x wmma into f32 accumulators.
// ---------------------------------------------------------------------------
__global__ __launch_bounds__(256) void flash_attn_kernel(
    const _Float16* __restrict__ qm, const _Float16* __restrict__ km,
    const _Float16* __restrict__ v, _Float16* __restrict__ o) {
  __shared__ __align__(16) _Float16 sKm[32 * NQ];     // 32 keys x 8 feat
  __shared__ __align__(16) _Float16 sVt[DK * 32];     // transposed [d][key]
  __shared__ __align__(16) _Float16 sP[8 * 16 * 32];  // per-wave P scratch

  const int t    = threadIdx.x;
  const int wave = t >> 5;
  const int lane = t & 31;
  const int hi   = lane >> 4;
  const int nl   = lane & 15;
  const int bh   = blockIdx.y;
  const int q0   = blockIdx.x * 128 + wave * 16;

  // Q fragment: real K only 0..7 -> lanes 0-15 carry data, rest zero.
  v16h a_q = zero_v16h();
  if (hi == 0) {
    _Float16 tmp[8];
    *(uint4*)tmp = *(const uint4*)(qm + ((size_t)bh * S + q0 + nl) * NQ);
#pragma unroll
    for (int e = 0; e < 8; ++e) a_q[e] = tmp[e];
  }

  v8f acc[4];
#pragma unroll
  for (int i = 0; i < 4; ++i) acc[i] = zero_v8f();
  float m_run[8], l_run[8];
#pragma unroll
  for (int r = 0; r < 8; ++r) { m_run[r] = -1e30f; l_run[r] = 0.0f; }

  for (int kb = 0; kb < S; kb += 32) {
    __syncthreads();
    // ---- cooperative staging shared by all 8 waves ----
    if (t < 32)  // Km: one 16B row per thread
      *(uint4*)(sKm + t * NQ) =
          *(const uint4*)(km + ((size_t)bh * S + kb + t) * NQ);
    {  // V: 32x64 f16 tile, stored transposed for contiguous B-frag loads
      const int key = t >> 3, c = t & 7;
      _Float16 tmp[8];
      *(uint4*)tmp =
          *(const uint4*)(v + (((size_t)bh * S + kb + key) * DK) + c * 8);
#pragma unroll
      for (int u = 0; u < 8; ++u) sVt[(c * 8 + u) * 32 + key] = tmp[u];
    }
    __syncthreads();

    // ---- scores: S[q, key] = sum_f Qm[q,f] Km[key,f]  (K padded 8->32) ----
    v16h b0 = zero_v16h(), b1 = zero_v16h();
    if (hi == 0) {
#pragma unroll
      for (int e = 0; e < 8; ++e) {
        b0[e] = sKm[nl * NQ + e];
        b1[e] = sKm[(16 + nl) * NQ + e];
      }
    }
    const v8f zc = zero_v8f();
    v8f s0 = __builtin_amdgcn_wmma_f32_16x16x32_f16(false, a_q, false, b0,
                                                    (short)0, zc, false, false);
    v8f s1 = __builtin_amdgcn_wmma_f32_16x16x32_f16(false, a_q, false, b1,
                                                    (short)0, zc, false, false);

    // ---- online softmax (rows live across a 16-lane half) ----
#pragma unroll
    for (int r = 0; r < 8; ++r) {
      float mx = fmaxf(s0[r], s1[r]);
#pragma unroll
      for (int off = 1; off < 16; off <<= 1)
        mx = fmaxf(mx, __shfl_xor(mx, off, 16));
      const float mnew  = fmaxf(m_run[r], mx);
      const float alpha = __expf(m_run[r] - mnew);
      m_run[r] = mnew;
      const float p0 = __expf(s0[r] - mnew);
      const float p1 = __expf(s1[r] - mnew);
      float rs = p0 + p1;
#pragma unroll
      for (int off = 1; off < 16; off <<= 1) rs += __shfl_xor(rs, off, 16);
      l_run[r] = l_run[r] * alpha + rs;
#pragma unroll
      for (int t4 = 0; t4 < 4; ++t4) acc[t4][r] *= alpha;
      const int rowp = r + 8 * hi;
      sP[wave * 512 + rowp * 32 + nl]      = (_Float16)p0;
      sP[wave * 512 + rowp * 32 + 16 + nl] = (_Float16)p1;
    }

    // wave-private LDS RAW across lanes: wait on the DS counter explicitly
    asm volatile("s_wait_dscnt 0" ::: "memory");

    // ---- P as A fragment (16 queries x 32 keys) ----
    v16h ap;
#pragma unroll
    for (int e = 0; e < 8; ++e)
      ap[e] = sP[wave * 512 + nl * 32 + hi * 16 + e];
#pragma unroll
    for (int e = 8; e < 16; ++e)
      ap[e] = sP[wave * 512 + nl * 32 + hi * 16 + e];

    // ---- PV: 4 wmma tiles across DK=64 ----
#pragma unroll
    for (int t4 = 0; t4 < 4; ++t4) {
      v16h bvf;
      const int d = t4 * 16 + nl;
#pragma unroll
      for (int e = 0; e < 16; ++e) bvf[e] = sVt[d * 32 + hi * 16 + e];
      acc[t4] = __builtin_amdgcn_wmma_f32_16x16x32_f16(false, ap, false, bvf,
                                                       (short)0, acc[t4],
                                                       false, false);
    }
  }

  // ---- normalize + write O as f16 in [B,S,E] (feeds final GEMM) ----
  const int b = bh >> 3, h = bh & 7;
#pragma unroll
  for (int r = 0; r < 8; ++r) {
    const float inv = 1.0f / l_run[r];
    const int row   = q0 + r + 8 * hi;
    const size_t base = ((size_t)(b * S + row)) * E + h * DK;
#pragma unroll
    for (int t4 = 0; t4 < 4; ++t4)
      o[base + t4 * 16 + nl] = (_Float16)(acc[t4][r] * inv);
  }
}

// ---------------------------------------------------------------------------
extern "C" void kernel_launch(void* const* d_in, const int* in_sizes, int n_in,
                              void* d_out, int out_size, void* d_ws,
                              size_t ws_size, hipStream_t stream) {
  (void)in_sizes; (void)n_in; (void)out_size; (void)ws_size;
  const float* x  = (const float*)d_in[0];
  const float* Wq = (const float*)d_in[1];
  const float* bq = (const float*)d_in[2];
  const float* Wk = (const float*)d_in[3];
  const float* bk = (const float*)d_in[4];
  const float* Wv = (const float*)d_in[5];
  const float* bv = (const float*)d_in[6];
  const float* Wo = (const float*)d_in[7];
  const float* bo = (const float*)d_in[8];
  // d_in[9] (kernel_params) cancels analytically: RZ commutes with Z.
  float* out = (float*)d_out;

  // workspace carve (18 MiB total)
  char* ws = (char*)d_ws;
  _Float16* v16 = (_Float16*)(ws);                                 // 8 MiB
  _Float16* qmb = (_Float16*)(ws + (size_t)8  * 1024 * 1024);      // 1 MiB
  _Float16* kmb = (_Float16*)(ws + (size_t)9  * 1024 * 1024);      // 1 MiB
  _Float16* o16 = (_Float16*)(ws + (size_t)10 * 1024 * 1024);      // 8 MiB

  const dim3 blk(256);
  gemm_proj<0><<<dim3(E / 128, (B * S) / 16), blk, 0, stream>>>(x, Wv, bv, v16);
  gemm_proj<1><<<dim3(1, (B * S) / 32), blk, 0, stream>>>(x, Wq, bq, qmb);
  gemm_proj<1><<<dim3(1, (B * S) / 32), blk, 0, stream>>>(x, Wk, bk, kmb);
  flash_attn_kernel<<<dim3(S / 128, B * H), blk, 0, stream>>>(qmb, kmb, v16, o16);
  gemm_proj<2><<<dim3(E / 128, (B * S) / 16), blk, 0, stream>>>(o16, Wo, bo, out);
}
